// RecursivePriorStateSpace_14396730376935
// MI455X (gfx1250) — matrix operand order repaired and checked by reading, not compiled, not run on hardware
//
#include <hip/hip_runtime.h>
#include <math.h>

typedef _Float16 v8h  __attribute__((ext_vector_type(8)));
typedef _Float16 v16h __attribute__((ext_vector_type(16)));
typedef float    v4f  __attribute__((ext_vector_type(4)));
typedef float    v8f  __attribute__((ext_vector_type(8)));

static constexpr int Bn   = 8;
static constexpr int Cn   = 256;   // input channels (== output channels)
static constexpr int HIDn = 256;   // hidden channels
static constexpr int Hn   = 128;
static constexpr int Wn   = 128;
static constexpr int HWn  = Hn * Wn;          // 16384 pixels per image
static constexpr int PIX  = Bn * HWn;         // 131072 pixels total
static constexpr int PXB  = 64;               // pixels per block
static constexpr int LROW = 48;               // LDS row stride in halves (96 B, 32B-aligned)

__device__ __forceinline__ v16h cat8(v8h lo, v8h hi) {
    return __builtin_shufflevector(lo, hi, 0,1,2,3,4,5,6,7,8,9,10,11,12,13,14,15);
}

// ---------------------------------------------------------------------------
// prep: f32 -> f16 weight copies; A = tanh(A_param)
// ---------------------------------------------------------------------------
__global__ void prep_kernel(const float* __restrict__ Wi, const float* __restrict__ Wo,
                            const float* __restrict__ Ap,
                            _Float16* __restrict__ Wi16, _Float16* __restrict__ Wo16,
                            float* __restrict__ At) {
    int t = blockIdx.x * 256 + threadIdx.x;
    if (t < HIDn * Cn) {
        Wi16[t] = (_Float16)Wi[t];
        Wo16[t] = (_Float16)Wo[t];
    }
    if (t < HIDn) At[t] = tanhf(Ap[t]);
}

// ---------------------------------------------------------------------------
// GEMM1: u[b,o,hw] = B_param[o] * (sum_c Wi[o,c]*(x[b,c,hw]+alpha*clip(prior)) + bi[o])
// Block: 256 threads (8 waves) handles 64 pixels x 256 HID outputs.
// ---------------------------------------------------------------------------
__global__ void __launch_bounds__(256)
gemm1_kernel(const float* __restrict__ x, const float* __restrict__ prior,
             const _Float16* __restrict__ Wi16, const float* __restrict__ bi,
             const float* __restrict__ Bp, const float* __restrict__ alpha_p,
             _Float16* __restrict__ u) {
    __shared__ __align__(32) _Float16 lds[PXB * LROW];

    const int pxg  = blockIdx.x * PXB;
    const int b    = pxg / HWn;
    const int hw   = pxg % HWn;
    const float alpha = *alpha_p;

    // staging role: thread -> (channel c in k-tile, pixel-group of 8)
    const int c   = threadIdx.x >> 3;
    const int pg  = threadIdx.x & 7;
    const int px0 = pg * 8;

    // alpha * clip(prior) for this thread's 8 pixels (invariant over K)
    float pa[8];
    #pragma unroll
    for (int i = 0; i < 8; ++i) {
        float p = prior[(size_t)b * HWn + hw + px0 + i];
        pa[i] = alpha * fminf(fmaxf(p, -1.0f), 1.0f);
    }

    // wave tiling: wave w -> M rows [w*32, w*32+32), 2 M-tiles x 4 N-tiles
    const int lane = threadIdx.x & 31;
    const int wv   = threadIdx.x >> 5;
    const int mb   = wv * 32;
    const int ksh  = (lane < 16) ? 0 : 16;   // A-fragment K half-offset
    const int bko  = (lane < 16) ? 0 : 16;   // B-fragment K offset (halves)
    const int rsh  = (lane < 16) ? 0 : 8;    // C/D row offset
    const int ln16 = lane & 15;

    v8f acc[2][4];
    #pragma unroll
    for (int mt = 0; mt < 2; ++mt)
        #pragma unroll
        for (int nt = 0; nt < 4; ++nt)
            acc[mt][nt] = {};

    for (int kb = 0; kb < Cn; kb += 32) {
        // ---- stage F_mod tile [32 ch x 64 px] into LDS as f16, px-major ----
        const float* xs = x + ((size_t)(b * Cn + kb + c)) * HWn + hw + px0;
        v4f f0 = *(const v4f*)xs;
        v4f f1 = *(const v4f*)(xs + 4);
        if (kb + 32 < Cn) {
            __builtin_prefetch(xs + (size_t)32 * HWn, 0, 1);   // next K-tile
        }
        #pragma unroll
        for (int i = 0; i < 4; ++i) {
            lds[(px0 + i    ) * LROW + c] = (_Float16)(f0[i] + pa[i]);
            lds[(px0 + i + 4) * LROW + c] = (_Float16)(f1[i] + pa[i + 4]);
        }
        __syncthreads();

        // ---- A fragments for the two M-tiles (from f16 Wi) ----
        v16h afrag[2];
        #pragma unroll
        for (int mt = 0; mt < 2; ++mt) {
            const _Float16* wp = Wi16 + (size_t)(mb + mt * 16 + ln16) * Cn + kb + ksh;
            afrag[mt] = *(const v16h*)wp;
        }
        // ---- 8 WMMAs: reuse each B fragment across both M-tiles ----
        #pragma unroll
        for (int nt = 0; nt < 4; ++nt) {
            v16h bfrag = *(const v16h*)(&lds[(nt * 16 + ln16) * LROW + bko]);
            #pragma unroll
            for (int mt = 0; mt < 2; ++mt) {
                acc[mt][nt] = __builtin_amdgcn_wmma_f32_16x16x32_f16(
                    false, afrag[mt], false, bfrag, (short)0, acc[mt][nt], false, false);
            }
        }
        __syncthreads();
    }

    // ---- epilogue: u = Bp[o] * (acc + bi[o]) -> f16 ----
    #pragma unroll
    for (int mt = 0; mt < 2; ++mt) {
        const int obase = mb + mt * 16 + rsh;
        v8f bi8 = *(const v8f*)(bi + obase);
        v8f bp8 = *(const v8f*)(Bp + obase);
        #pragma unroll
        for (int nt = 0; nt < 4; ++nt) {
            _Float16* up = u + (size_t)b * HIDn * HWn + hw + nt * 16 + ln16;
            #pragma unroll
            for (int v = 0; v < 8; ++v)
                up[(size_t)(obase + v) * HWn] =
                    (_Float16)((acc[mt][nt][v] + bi8[v]) * bp8[v]);
        }
    }
}

// ---------------------------------------------------------------------------
// Horizontal scan: one thread per (b,o,h) row; h_t = A*h_{t-1} + u_t along W.
// Vectorized: 16 x (load v8h, 8 chained FMAs, store v8h). u/hbuf are
// L2-resident (134 MB < 192 MB L2), so this is issue-bound -> 16B per access.
// ---------------------------------------------------------------------------
__global__ void scan_h_kernel(const _Float16* __restrict__ u, const float* __restrict__ At,
                              _Float16* __restrict__ hb) {
    int idx = blockIdx.x * 256 + threadIdx.x;        // [0, B*HID*H)
    int h = idx % Hn;
    int o = (idx / Hn) % HIDn;
    int b = idx / (Hn * HIDn);
    const float A = At[o];
    size_t base = ((size_t)(b * HIDn + o)) * HWn + (size_t)h * Wn;
    float acc = 0.0f;
    for (int w8 = 0; w8 < Wn / 8; ++w8) {
        v8h uv = *(const v8h*)(u + base + w8 * 8);
        v8h r;
        #pragma unroll
        for (int i = 0; i < 8; ++i) {
            acc = fmaf(A, acc, (float)uv[i]);
            r[i] = (_Float16)acc;
        }
        *(v8h*)(hb + base + w8 * 8) = r;
    }
}

// ---------------------------------------------------------------------------
// Vertical scan: each thread owns 8 adjacent columns (8 independent
// accumulators -> hides recurrence latency; 16B/lane fully-coalesced steps).
// Fuses h_fused = h_h + h_v into hbuf.
// ---------------------------------------------------------------------------
__global__ void scan_v_kernel(const _Float16* __restrict__ u, const float* __restrict__ At,
                              _Float16* __restrict__ hb) {
    int idx = blockIdx.x * 256 + threadIdx.x;        // [0, B*HID*W/8)
    int w8 = idx % (Wn / 8);
    int o  = (idx / (Wn / 8)) % HIDn;
    int b  = idx / ((Wn / 8) * HIDn);
    const float A = At[o];
    size_t base = ((size_t)(b * HIDn + o)) * HWn + (size_t)w8 * 8;
    v8f acc = {};
    for (int h = 0; h < Hn; ++h) {
        size_t i = base + (size_t)h * Wn;
        v8h uv = *(const v8h*)(u + i);
        v8h hv = *(const v8h*)(hb + i);
        v8h r;
        #pragma unroll
        for (int j = 0; j < 8; ++j) {
            acc[j] = fmaf(A, acc[j], (float)uv[j]);
            r[j] = (_Float16)((float)hv[j] + acc[j]);
        }
        *(v8h*)(hb + i) = r;
    }
}

// ---------------------------------------------------------------------------
// GEMM2: out = F_mod + gamma * (Wo @ h_fused + bo)
// ---------------------------------------------------------------------------
__global__ void __launch_bounds__(256)
gemm2_kernel(const _Float16* __restrict__ hb, const _Float16* __restrict__ Wo16,
             const float* __restrict__ bo, const float* __restrict__ x,
             const float* __restrict__ prior, const float* __restrict__ alpha_p,
             const float* __restrict__ gamma_p, float* __restrict__ out) {
    __shared__ __align__(32) _Float16 lds[PXB * LROW];

    const int pxg = blockIdx.x * PXB;
    const int b   = pxg / HWn;
    const int hw  = pxg % HWn;
    const float alpha = *alpha_p;
    const float gamma = *gamma_p;

    const int c   = threadIdx.x >> 3;
    const int pg  = threadIdx.x & 7;
    const int px0 = pg * 8;

    const int lane = threadIdx.x & 31;
    const int wv   = threadIdx.x >> 5;
    const int mb   = wv * 32;
    const int ksh  = (lane < 16) ? 0 : 16;
    const int bko  = (lane < 16) ? 0 : 16;
    const int rsh  = (lane < 16) ? 0 : 8;
    const int ln16 = lane & 15;

    v8f acc[2][4];
    #pragma unroll
    for (int mt = 0; mt < 2; ++mt)
        #pragma unroll
        for (int nt = 0; nt < 4; ++nt)
            acc[mt][nt] = {};

    for (int kb = 0; kb < HIDn; kb += 32) {
        // stage h_fused tile [32 hid x 64 px] (already f16)
        const _Float16* hs = hb + ((size_t)(b * HIDn + kb + c)) * HWn + hw + px0;
        v8h hv = *(const v8h*)hs;
        if (kb + 32 < HIDn) {
            __builtin_prefetch(hs + (size_t)32 * HWn, 0, 1);   // next K-tile
        }
        #pragma unroll
        for (int i = 0; i < 8; ++i)
            lds[(px0 + i) * LROW + c] = hv[i];
        __syncthreads();

        v16h afrag[2];
        #pragma unroll
        for (int mt = 0; mt < 2; ++mt) {
            const _Float16* wp = Wo16 + (size_t)(mb + mt * 16 + ln16) * HIDn + kb + ksh;
            afrag[mt] = *(const v16h*)wp;
        }
        #pragma unroll
        for (int nt = 0; nt < 4; ++nt) {
            v16h bfrag = *(const v16h*)(&lds[(nt * 16 + ln16) * LROW + bko]);
            #pragma unroll
            for (int mt = 0; mt < 2; ++mt) {
                acc[mt][nt] = __builtin_amdgcn_wmma_f32_16x16x32_f16(
                    false, afrag[mt], false, bfrag, (short)0, acc[mt][nt], false, false);
            }
        }
        __syncthreads();
    }

    // epilogue: out = x + alpha*clip(prior) + gamma*(acc + bo)
    float pr[4];
    #pragma unroll
    for (int nt = 0; nt < 4; ++nt) {
        float p = prior[(size_t)b * HWn + hw + nt * 16 + ln16];
        pr[nt] = alpha * fminf(fmaxf(p, -1.0f), 1.0f);
    }
    #pragma unroll
    for (int mt = 0; mt < 2; ++mt) {
        const int obase = mb + mt * 16 + rsh;
        v8f bo8 = *(const v8f*)(bo + obase);
        #pragma unroll
        for (int nt = 0; nt < 4; ++nt) {
            const int px = nt * 16 + ln16;
            #pragma unroll
            for (int v = 0; v < 8; ++v) {
                size_t oi = ((size_t)(b * Cn + obase + v)) * HWn + hw + px;
                out[oi] = x[oi] + pr[nt] + gamma * (acc[mt][nt][v] + bo8[v]);
            }
        }
    }
}

// ---------------------------------------------------------------------------
extern "C" void kernel_launch(void* const* d_in, const int* in_sizes, int n_in,
                              void* d_out, int out_size, void* d_ws, size_t ws_size,
                              hipStream_t stream) {
    const float* x     = (const float*)d_in[0];
    const float* prior = (const float*)d_in[1];
    const float* Wi    = (const float*)d_in[2];
    const float* bi    = (const float*)d_in[3];
    const float* Ap    = (const float*)d_in[4];
    const float* Bp    = (const float*)d_in[5];
    const float* alpha = (const float*)d_in[6];
    const float* gamma = (const float*)d_in[7];
    const float* Wo    = (const float*)d_in[8];
    const float* bo    = (const float*)d_in[9];
    float* out = (float*)d_out;

    // workspace carving (all offsets 256B aligned)
    char* ws = (char*)d_ws;
    const size_t uBytes = (size_t)Bn * HIDn * HWn * sizeof(_Float16);  // 64 MiB
    _Float16* Wi16 = (_Float16*)(ws);                      // 128 KB
    _Float16* Wo16 = (_Float16*)(ws + 131072);             // 128 KB
    float*    At   = (float*)   (ws + 262144);             // 1 KB
    _Float16* u    = (_Float16*)(ws + 263168);             // u  [B,HID,H,W] f16
    _Float16* hbuf = (_Float16*)(ws + 263168 + uBytes);    // h_fused, same shape

    prep_kernel<<<(HIDn * Cn + 255) / 256, 256, 0, stream>>>(Wi, Wo, Ap, Wi16, Wo16, At);

    gemm1_kernel<<<PIX / PXB, 256, 0, stream>>>(x, prior, Wi16, bi, Bp, alpha, u);

    scan_h_kernel<<<(Bn * HIDn * Hn) / 256, 256, 0, stream>>>(u, At, hbuf);
    scan_v_kernel<<<(Bn * HIDn * (Wn / 8)) / 256, 256, 0, stream>>>(u, At, hbuf);

    gemm2_kernel<<<PIX / PXB, 256, 0, stream>>>(hbuf, Wo16, bo, x, prior, alpha, gamma, out);
}
